// AlternateGCN_66537633350122
// MI455X (gfx1250) — compile-verified, matrix operand order checked by reading, not run on hardware
//
#include <hip/hip_runtime.h>

// ---------------------------------------------------------------------------
// CDNA5 (gfx1250) GCN + edge-MLP pipeline.
// wave32; WMMA bf16 16x16x32 with f32 accumulate for every matmul.
// Node features (25.6MB) are L2-resident on MI455X (192MB L2), so the edge
// gathers are cheap; the decoder is fully fused per 16-edge tile so the
// [E,256]/[E,128] intermediates (2.5GB of would-be HBM traffic) stay in LDS.
// The decoder additionally prefetches the NEXT tile's gathered node rows
// (global_prefetch_b8) so the random L2 gather latency overlaps the WMMAs.
// ---------------------------------------------------------------------------

typedef __attribute__((ext_vector_type(16))) __bf16 v16bf;
typedef __attribute__((ext_vector_type(8)))  float  v8f;

union ABFrag { v16bf v; uint4 q[2]; };

// f32 -> bf16 flat convert
__global__ void f32_to_bf16_kernel(const float* __restrict__ in,
                                   __bf16* __restrict__ out, long long n) {
    long long i = (long long)blockIdx.x * blockDim.x + threadIdx.x;
    if (i < n) out[i] = (__bf16)in[i];
}

// f32 [R][C] row-major -> bf16 [C][R] (transposed) so B-fragments load
// contiguously (lane holds col, K runs along memory).
__global__ void transpose_bf16_kernel(const float* __restrict__ in,
                                      __bf16* __restrict__ out, int R, int C) {
    int idx = blockIdx.x * blockDim.x + threadIdx.x;
    if (idx < R * C) {
        int r = idx / C, c = idx % C;
        out[(size_t)c * R + r] = (__bf16)in[idx];
    }
}

// deg[col[e]] += (w ? w[e] : 1)
__global__ void deg_kernel(const int* __restrict__ cols,
                           const float* __restrict__ w,
                           float* __restrict__ deg, int E) {
    int i = blockIdx.x * blockDim.x + threadIdx.x;
    if (i < E) atomicAdd(&deg[cols[i]], w ? w[i] : 1.0f);
}

__global__ void dinv_kernel(const float* __restrict__ deg,
                            float* __restrict__ dinv, int n) {
    int i = blockIdx.x * blockDim.x + threadIdx.x;
    if (i < n) { float d = deg[i]; dinv[i] = d > 0.f ? rsqrtf(d) : 0.f; }
}

// C[M,Nc] = (gather? A[gather[m]] : A[m]) @ Bt^T ; A bf16 [*,K], Bt bf16 [Nc][K].
// One wave per 16x16 tile; 8 waves/block cover 128 columns.
__global__ __launch_bounds__(256) void wmma_gemm_kernel(
    const __bf16* __restrict__ A, const int* __restrict__ gather,
    const __bf16* __restrict__ Bt, float* __restrict__ C,
    int M, int K, int Nc) {
    const int t = threadIdx.x, wv = t >> 5, lane = t & 31;
    const int half = (lane >> 4) & 1, lr = lane & 15;
    const int m0 = blockIdx.x * 16;
    const int n0 = blockIdx.y * 128 + wv * 16;
    if (n0 >= Nc) return;                 // wave-uniform; EXEC stays all-1s
    int gm = m0 + lr; if (gm >= M) gm = M - 1;
    const int arow = gather ? gather[gm] : gm;
    const __bf16* ap = A + (size_t)arow * K;
    const __bf16* bp = Bt + (size_t)(n0 + lr) * K;
    v8f acc = {};
    for (int kk = 0; kk < K; kk += 32) {
        ABFrag a, b;
        const int k0 = kk + half * 8;     // two contiguous 8-elem K chunks
        a.q[0] = *(const uint4*)(ap + k0);
        a.q[1] = *(const uint4*)(ap + k0 + 16);
        b.q[0] = *(const uint4*)(bp + k0);
        b.q[1] = *(const uint4*)(bp + k0 + 16);
        acc = __builtin_amdgcn_wmma_f32_16x16x32_bf16(
            false, a.v, false, b.v, (short)0, acc, false, false);
    }
#pragma unroll
    for (int r = 0; r < 8; ++r) {
        const int row = m0 + r + half * 8;
        if (row < M) C[(size_t)row * Nc + n0 + lr] = acc[r];
    }
}

// agg[col[e]] += dinv[row]*dinv[col]*(w?w[e]:1) * h[row]; float4 per thread.
__global__ void scatter_kernel(const float* __restrict__ h,
                               const int* __restrict__ rows,
                               const int* __restrict__ cols,
                               const float* __restrict__ w,
                               const float* __restrict__ dinv,
                               float* __restrict__ agg, int E, int C) {
    long long idx = (long long)blockIdx.x * blockDim.x + threadIdx.x;
    const int cpt = C >> 2;
    if (idx >= (long long)E * cpt) return;
    const int e  = (int)(idx / cpt);
    const int c4 = (int)(idx % cpt) << 2;
    const int r = rows[e], c = cols[e];
    const float norm = dinv[r] * dinv[c] * (w ? w[e] : 1.0f);
    const float4 v = *(const float4*)(h + (size_t)r * C + c4);
    float* d = agg + (size_t)c * C + c4;
    atomicAdd(d + 0, v.x * norm);
    atomicAdd(d + 1, v.y * norm);
    atomicAdd(d + 2, v.z * norm);
    atomicAdd(d + 3, v.w * norm);
}

// out_bf16 = ELU(agg + bias[c])
__global__ void bias_elu_kernel(const float* __restrict__ agg,
                                const float* __restrict__ bias,
                                __bf16* __restrict__ out, long long n, int C) {
    long long i = (long long)blockIdx.x * blockDim.x + threadIdx.x;
    if (i < n) {
        float v = agg[i] + bias[(int)(i % C)];
        v = v > 0.f ? v : (__expf(v) - 1.f);
        out[i] = (__bf16)v;
    }
}

// Fused edge decoder: per 16-edge tile:
//   A = concat(x[src], x[dst]) [16,256] -> LDS,
//   h = relu(A@W1+b1) [16,128] (WMMA, K=256) -> LDS bf16,
//   h = relu(h@W2+b2) [16,128] (WMMA, K=128) -> LDS f32,
//   out = h@w3 + b3 (LDS dot-reduce).
__global__ __launch_bounds__(256) void decoder_kernel(
    const __bf16* __restrict__ x2b, const int* __restrict__ src,
    const int* __restrict__ dst, const __bf16* __restrict__ W1t,
    const float* __restrict__ b1, const __bf16* __restrict__ W2t,
    const float* __restrict__ b2, const float* __restrict__ w3,
    const float* __restrict__ b3, float* __restrict__ out, int E) {
    __shared__ __bf16 ldsA[16][256];
    __shared__ __bf16 hA[16][128];
    __shared__ float  h2f[16][128];
    __shared__ float  outAcc[16];
    const int t = threadIdx.x;
    const int e0 = blockIdx.x * 16;
    if (t < 16) outAcc[t] = 0.0f;
    {   // stage gathered+concat features into LDS (32B per thread)
        const int e = t >> 4, part = t & 15;
        int edge = e0 + e; if (edge >= E) edge = E - 1;
        const int node = (part < 8) ? src[edge] : dst[edge];
        const int koff = (part & 7) * 16;
        const uint4* s = (const uint4*)(x2b + (size_t)node * 128 + koff);
        uint4* d = (uint4*)(&ldsA[e][((part < 8) ? 0 : 128) + koff]);
        d[0] = s[0]; d[1] = s[1];
        // Prefetch the NEXT tile's gathered rows into cache while this
        // tile's 12 WMMAs run (lowers to global_prefetch_b8 on gfx1250).
        const int edge2 = e0 + 16 + e;
        if (edge2 < E) {
            const int node2 = (part < 8) ? src[edge2] : dst[edge2];
            __builtin_prefetch(x2b + (size_t)node2 * 128 + koff, 0, 0);
        }
    }
    __syncthreads();
    const int wv = t >> 5, lane = t & 31;
    const int half = (lane >> 4) & 1, lr = lane & 15;
    {   // GEMM1: [16,256] x W1 -> relu -> hA (bf16)
        v8f acc = {};
#pragma unroll
        for (int kk = 0; kk < 256; kk += 32) {
            ABFrag a, b;
            const int k0 = kk + half * 8;
            a.q[0] = *(const uint4*)(&ldsA[lr][k0]);
            a.q[1] = *(const uint4*)(&ldsA[lr][k0 + 16]);
            const __bf16* bp = W1t + (size_t)(wv * 16 + lr) * 256;
            b.q[0] = *(const uint4*)(bp + k0);
            b.q[1] = *(const uint4*)(bp + k0 + 16);
            acc = __builtin_amdgcn_wmma_f32_16x16x32_bf16(
                false, a.v, false, b.v, (short)0, acc, false, false);
        }
        const float bias = b1[wv * 16 + lr];
#pragma unroll
        for (int r = 0; r < 8; ++r) {
            float v = acc[r] + bias;
            hA[r + half * 8][wv * 16 + lr] = (__bf16)(v > 0.f ? v : 0.f);
        }
    }
    __syncthreads();
    {   // GEMM2: [16,128] x W2 -> relu -> h2f (f32)
        v8f acc = {};
#pragma unroll
        for (int kk = 0; kk < 128; kk += 32) {
            ABFrag a, b;
            const int k0 = kk + half * 8;
            a.q[0] = *(const uint4*)(&hA[lr][k0]);
            a.q[1] = *(const uint4*)(&hA[lr][k0 + 16]);
            const __bf16* bp = W2t + (size_t)(wv * 16 + lr) * 128;
            b.q[0] = *(const uint4*)(bp + k0);
            b.q[1] = *(const uint4*)(bp + k0 + 16);
            acc = __builtin_amdgcn_wmma_f32_16x16x32_bf16(
                false, a.v, false, b.v, (short)0, acc, false, false);
        }
        const float bias = b2[wv * 16 + lr];
#pragma unroll
        for (int r = 0; r < 8; ++r) {
            float v = acc[r] + bias;
            h2f[r + half * 8][wv * 16 + lr] = v > 0.f ? v : 0.f;
        }
    }
    __syncthreads();
    {   // final dot with w3 (16 threads per edge, 8 elems each)
        const int e = t >> 4, seg = t & 15;
        float p = 0.f;
#pragma unroll
        for (int k = 0; k < 8; ++k) p += h2f[e][seg * 8 + k] * w3[seg * 8 + k];
        atomicAdd(&outAcc[e], p);
    }
    __syncthreads();
    if (t < 16 && e0 + t < E) out[e0 + t] = outAcc[t] + b3[0];
}

extern "C" void kernel_launch(void* const* d_in, const int* in_sizes, int n_in,
                              void* d_out, int out_size, void* d_ws, size_t ws_size,
                              hipStream_t stream) {
    const int*   node_ids  = (const int*)d_in[0];
    const int*   edge_idx  = (const int*)d_in[1];
    const int*   nbr_idx   = (const int*)d_in[2];
    const float* edge_attr = (const float*)d_in[3];
    const float* emb   = (const float*)d_in[4];
    const float* W_in  = (const float*)d_in[5];
    const float* b_in  = (const float*)d_in[6];
    const float* W_out = (const float*)d_in[7];
    const float* b_out = (const float*)d_in[8];
    const float* W1    = (const float*)d_in[9];
    const float* b1    = (const float*)d_in[10];
    const float* W2    = (const float*)d_in[11];
    const float* b2    = (const float*)d_in[12];
    const float* W3    = (const float*)d_in[13];
    const float* b3    = (const float*)d_in[14];

    const int N = in_sizes[0];
    const int E = in_sizes[3];
    const int D = 128, H = 256;
    const int* src  = edge_idx;       const int* dstv = edge_idx + E;
    const int* nsrc = nbr_idx;        const int* ndst = nbr_idx + E;

    // -------- workspace layout (aliased; ~155 MB total) --------
    char* ws = (char*)d_ws;
    size_t off = 0;
    auto alloc = [&](size_t bytes) -> char* {
        char* p = ws + off; off += (bytes + 255) & ~(size_t)255; return p;
    };
    __bf16* embb   = (__bf16*)alloc((size_t)N * D * 2);
    __bf16* W_int  = (__bf16*)alloc((size_t)H * D * 2);   // [256][128]
    __bf16* W_outt = (__bf16*)alloc((size_t)D * H * 2);   // [128][256]
    __bf16* W1t    = (__bf16*)alloc((size_t)D * 2 * D * 2); // [128][256]
    __bf16* W2t    = (__bf16*)alloc((size_t)D * D * 2);   // [128][128]
    float*  deg1   = (float*)alloc((size_t)N * 4);
    float*  deg2   = (float*)alloc((size_t)N * 4);
    float*  dinv1  = (float*)alloc((size_t)N * 4);
    float*  dinv2  = (float*)alloc((size_t)N * 4);
    float*  bufA   = (float*)alloc((size_t)N * H * 4);    // h1, then agg2
    float*  bufB   = (float*)alloc((size_t)N * H * 4);    // agg1, then h2
    __bf16* x1b    = (__bf16*)alloc((size_t)N * H * 2);
    __bf16* x2b    = (__bf16*)alloc((size_t)N * D * 2);
    float* h1 = bufA; float* agg2 = bufA;   // h1 dead before agg2 is written
    float* agg1 = bufB; float* h2 = bufB;   // agg1 dead before h2 is written
    (void)ws_size; (void)n_in; (void)out_size;

    // -------- weight prep: convert + transpose to bf16 --------
    const long long nemb = (long long)N * D;
    f32_to_bf16_kernel<<<(unsigned)((nemb + 255) / 256), 256, 0, stream>>>(emb, embb, nemb);
    transpose_bf16_kernel<<<(D * H + 255) / 256, 256, 0, stream>>>(W_in, W_int, D, H);
    transpose_bf16_kernel<<<(H * D + 255) / 256, 256, 0, stream>>>(W_out, W_outt, H, D);
    transpose_bf16_kernel<<<(2 * D * D + 255) / 256, 256, 0, stream>>>(W1, W1t, 2 * D, D);
    transpose_bf16_kernel<<<(D * D + 255) / 256, 256, 0, stream>>>(W2, W2t, D, D);

    // -------- degrees / norms --------
    hipMemsetAsync(deg1, 0, (size_t)N * 4, stream);
    hipMemsetAsync(deg2, 0, (size_t)N * 4, stream);
    hipMemsetAsync(agg1, 0, (size_t)N * H * 4, stream);
    deg_kernel<<<(E + 255) / 256, 256, 0, stream>>>(dstv, edge_attr, deg1, E);
    deg_kernel<<<(E + 255) / 256, 256, 0, stream>>>(ndst, nullptr, deg2, E);
    dinv_kernel<<<(N + 255) / 256, 256, 0, stream>>>(deg1, dinv1, N);
    dinv_kernel<<<(N + 255) / 256, 256, 0, stream>>>(deg2, dinv2, N);

    // -------- GCN conv 1 --------
    dim3 g1((N + 15) / 16, H / 128);
    wmma_gemm_kernel<<<g1, 256, 0, stream>>>(embb, node_ids, W_int, h1, N, D, H);
    long long sc1 = (long long)E * (H / 4);
    scatter_kernel<<<(unsigned)((sc1 + 255) / 256), 256, 0, stream>>>(
        h1, src, dstv, edge_attr, dinv1, agg1, E, H);
    long long na1 = (long long)N * H;
    bias_elu_kernel<<<(unsigned)((na1 + 255) / 256), 256, 0, stream>>>(agg1, b_in, x1b, na1, H);

    // -------- GCN conv 2 --------
    dim3 g2((N + 15) / 16, 1);
    wmma_gemm_kernel<<<g2, 256, 0, stream>>>(x1b, nullptr, W_outt, h2, N, H, D);
    hipMemsetAsync(agg2, 0, (size_t)N * D * 4, stream);
    long long sc2 = (long long)E * (D / 4);
    scatter_kernel<<<(unsigned)((sc2 + 255) / 256), 256, 0, stream>>>(
        h2, nsrc, ndst, nullptr, dinv2, agg2, E, D);
    long long na2 = (long long)N * D;
    bias_elu_kernel<<<(unsigned)((na2 + 255) / 256), 256, 0, stream>>>(agg2, b_out, x2b, na2, D);

    // -------- fused edge decoder --------
    decoder_kernel<<<(E + 15) / 16, 256, 0, stream>>>(
        x2b, src, dstv, W1t, b1, W2t, b2, W3, b3, (float*)d_out, E);
}
